// Blocks_31748398252103
// MI455X (gfx1250) — matrix operand order repaired and checked by reading, not compiled, not run on hardware
//
#include <hip/hip_runtime.h>

// ---------------------------------------------------------------------------
// Transformer block forward for MI455X (gfx1250, wave32, WMMA).
// All matmuls via v_wmma_f32_16x16x32_bf16 (f32 accum). Flash-style causal
// attention with online softmax. LayerNorm / residuals in fp32.
// GEMM: 32x64 wave tile, software-pipelined (double-buffered operands) so
// WMMAs overlap the next k-step's global_load_b128 clauses.
// ---------------------------------------------------------------------------

typedef __bf16 bf16;
typedef __attribute__((ext_vector_type(8)))  __bf16 v8bf;
typedef __attribute__((ext_vector_type(16))) __bf16 v16bf;
typedef __attribute__((ext_vector_type(8)))  float  v8f;

#define E_DIM  1024
#define S_LEN  2048
#define H_NUM  16
#define DH     64
#define FF_DIM 4096
#define M_ROWS 4096   // B * S

static __device__ __forceinline__ v8bf ld8(const bf16* p) {
  return *(const v8bf*)p;
}
static __device__ __forceinline__ v16bf cat8(v8bf lo, v8bf hi) {
  return __builtin_shufflevector(lo, hi, 0,1,2,3,4,5,6,7,8,9,10,11,12,13,14,15);
}
static __device__ __forceinline__ v8f wmma_bf16(v16bf a, v16bf b, v8f c) {
  // D = A(16x32 bf16) * B(32x16 bf16) + C(16x16 f32)
  return __builtin_amdgcn_wmma_f32_16x16x32_bf16(
      /*neg_a=*/false, a, /*neg_b=*/false, b,
      /*c_mod=*/(short)0, c, /*reuse_a=*/false, /*reuse_b=*/false);
}
static __device__ __forceinline__ float half_reduce_max(float v) {
  v = fmaxf(v, __shfl_xor(v, 1));
  v = fmaxf(v, __shfl_xor(v, 2));
  v = fmaxf(v, __shfl_xor(v, 4));
  v = fmaxf(v, __shfl_xor(v, 8));
  return v;
}
static __device__ __forceinline__ float half_reduce_sum(float v) {
  v += __shfl_xor(v, 1);
  v += __shfl_xor(v, 2);
  v += __shfl_xor(v, 4);
  v += __shfl_xor(v, 8);
  return v;
}

// --------------------------- weight conversion -----------------------------
// Wq/Wk: [H][E][Dh] fp32 -> WT[n = h*64+d][e] bf16 (row-major by output col)
__global__ __launch_bounds__(256)
void cvt_qkw_kernel(const float* __restrict__ W, bf16* __restrict__ WT) {
  const int idx = blockIdx.x * 256 + threadIdx.x;           // over 1M
  const int n = idx >> 10, e = idx & (E_DIM - 1);
  WT[idx] = (bf16)W[((size_t)(n >> 6) * E_DIM + e) * DH + (n & 63)];
}

// W[K][N] fp32 -> WT[N][K] bf16
__global__ __launch_bounds__(256)
void cvt_T_kernel(const float* __restrict__ W, bf16* __restrict__ WT,
                  int K, int N) {
  const size_t idx = (size_t)blockIdx.x * 256 + threadIdx.x; // grid exact
  const size_t n = idx / (size_t)K, k = idx % (size_t)K;
  WT[idx] = (bf16)W[k * (size_t)N + n];
}

// ------------------------------- LayerNorm ---------------------------------
// one block (256 thr) per row of 1024; fp32 in, bf16 out
__global__ __launch_bounds__(256)
void ln_kernel(const float* __restrict__ x, const float* __restrict__ g,
               const float* __restrict__ beta, bf16* __restrict__ outH) {
  __shared__ float sred[8], ssred[8];
  const int row = blockIdx.x;
  const int tid = threadIdx.x;
  const float4 v = ((const float4*)(x + (size_t)row * E_DIM))[tid];
  float s  = v.x + v.y + v.z + v.w;
  float ss = v.x * v.x + v.y * v.y + v.z * v.z + v.w * v.w;
#pragma unroll
  for (int off = 1; off < 32; off <<= 1) {
    s += __shfl_xor(s, off);
    ss += __shfl_xor(ss, off);
  }
  const int lane = tid & 31, wv = tid >> 5;
  if (lane == 0) { sred[wv] = s; ssred[wv] = ss; }
  __syncthreads();
  s = 0.f; ss = 0.f;
#pragma unroll
  for (int i = 0; i < 8; ++i) { s += sred[i]; ss += ssred[i]; }
  const float mu   = s * (1.0f / E_DIM);
  const float var  = ss * (1.0f / E_DIM) - mu * mu;
  const float rstd = rsqrtf(var + 1e-5f);
  const int c = tid * 4;
  bf16* o = outH + (size_t)row * E_DIM + c;
  o[0] = (bf16)((v.x - mu) * rstd * g[c + 0] + beta[c + 0]);
  o[1] = (bf16)((v.y - mu) * rstd * g[c + 1] + beta[c + 1]);
  o[2] = (bf16)((v.z - mu) * rstd * g[c + 2] + beta[c + 2]);
  o[3] = (bf16)((v.w - mu) * rstd * g[c + 3] + beta[c + 3]);
}

// ------------------------------ WMMA GEMM ----------------------------------
// C[M][N] = A[M][K](bf16,row-major) * BT[N][K](bf16,row-major)^T
//           [+bias[n]] [+resid[m][n] f32] [relu]
// Wave tile 32x64: 2 A operands x 4 B operands -> 8 WMMAs per k-step.
// Operands double-buffered: next k-step's 12 b128 loads issue before the
// current WMMAs, so the loadcnt wait at loop top is already satisfied.
__global__ __launch_bounds__(256)
void gemm_bf16_kernel(const bf16* __restrict__ A, const bf16* __restrict__ BT,
                      const float* __restrict__ bias,
                      const float* __restrict__ resid,
                      float* __restrict__ outF, bf16* __restrict__ outH,
                      bf16* __restrict__ outT,
                      int M, int N, int K, int do_relu) {
  const int lane   = threadIdx.x & 31;
  const int laneLo = lane & 15, laneHi = lane >> 4;
  const int wid    = blockIdx.x * (blockDim.x >> 5) + (threadIdx.x >> 5);
  const int ntiles = N >> 6;
  const int mt = wid / ntiles, nt = wid % ntiles;
  const int m0 = mt << 5, n0 = nt << 6;
  if (m0 >= M) return;

  // A operand rows (K-interleave via laneHi folded into the base)
  const bf16* arow0 = A + (size_t)(m0 + laneLo) * K + 8 * laneHi;
  const bf16* arow1 = arow0 + (size_t)16 * K;
  const bf16* brow[4];
#pragma unroll
  for (int t = 0; t < 4; ++t)
    brow[t] = BT + (size_t)(n0 + t * 16 + laneLo) * K + 8 * laneHi;

  v8f acc[8] = {};

  // preload k = 0
  v16bf a0 = cat8(ld8(arow0), ld8(arow0 + 16));
  v16bf a1 = cat8(ld8(arow1), ld8(arow1 + 16));
  v16bf bb[4];
#pragma unroll
  for (int t = 0; t < 4; ++t)
    bb[t] = cat8(ld8(brow[t]), ld8(brow[t] + 16));

  for (int k0 = 0; k0 < K; k0 += 32) {
    int kn = k0 + 32;
    if (kn >= K) kn = 0;               // last-iter prefetch: harmless re-read
    // prefetch next k-step into distinct registers
    v16bf na0 = cat8(ld8(arow0 + kn), ld8(arow0 + kn + 16));
    v16bf na1 = cat8(ld8(arow1 + kn), ld8(arow1 + kn + 16));
    v16bf nb[4];
#pragma unroll
    for (int t = 0; t < 4; ++t)
      nb[t] = cat8(ld8(brow[t] + kn), ld8(brow[t] + kn + 16));

    // compute with current operands (overlaps the prefetch above)
#pragma unroll
    for (int t = 0; t < 4; ++t) acc[t]     = wmma_bf16(a0, bb[t], acc[t]);
#pragma unroll
    for (int t = 0; t < 4; ++t) acc[4 + t] = wmma_bf16(a1, bb[t], acc[4 + t]);

    a0 = na0; a1 = na1;
#pragma unroll
    for (int t = 0; t < 4; ++t) bb[t] = nb[t];
  }

#pragma unroll
  for (int half = 0; half < 2; ++half) {
#pragma unroll
    for (int t = 0; t < 4; ++t) {
#pragma unroll
      for (int i = 0; i < 8; ++i) {
        const int m = m0 + half * 16 + i + 8 * laneHi;
        const int n = n0 + t * 16 + laneLo;
        float v = acc[half * 4 + t][i];
        if (bias)    v += bias[n];
        if (resid)   v += resid[(size_t)m * N + n];
        if (do_relu) v = fmaxf(v, 0.0f);
        if (outF) outF[(size_t)m * N + n] = v;
        if (outH) outH[(size_t)m * N + n] = (bf16)v;
        if (outT) {   // [b][h][d][s] for PV B-operand (M=4096, N=1024 only)
          const int b = m >> 11, s = m & (S_LEN - 1);
          const int hh = n >> 6, d = n & 63;
          outT[(((size_t)(b * H_NUM + hh) * DH + d) << 11) + s] = (bf16)v;
        }
      }
    }
  }
}

// ------------------------- causal flash attention --------------------------
// wave = one 16-query tile of one (b,h). V == K (faithful reference bug).
__global__ __launch_bounds__(128)
void attn_kernel(const bf16* __restrict__ Qb, const bf16* __restrict__ Kb,
                 const bf16* __restrict__ Ktb, bf16* __restrict__ Ob) {
  __shared__ __align__(16) bf16 Plds[4][16 * 32];
  const int lane   = threadIdx.x & 31;
  const int wv     = threadIdx.x >> 5;
  const int laneLo = lane & 15, laneHi = lane >> 4;
  const int tile = blockIdx.x * 4 + wv;       // B*H*(S/16) = 4096 tiles
  const int qt = tile & 127;
  const int bh = tile >> 7;                   // b*16 + h
  const int q0 = qt << 4;
  const int b = bh >> 4, h = bh & 15;

  // Q tile as two A-operands (Dh = 64 = 2 x K32)
  const bf16* qrow = Qb + (size_t)(b * S_LEN + q0 + laneLo) * E_DIM + h * DH;
  const v16bf aq0 = cat8(ld8(qrow +      16 * laneHi), ld8(qrow + 8 + 16 * laneHi));
  const v16bf aq1 = cat8(ld8(qrow + 32 + 16 * laneHi), ld8(qrow + 40 + 16 * laneHi));

  v8f o[4] = {};
  float mrow[8], lrow[8];
#pragma unroll
  for (int i = 0; i < 8; ++i) { mrow[i] = -3.0e38f; lrow[i] = 0.0f; }

  bf16* P = &Plds[wv][0];
  const bf16* vbase = Ktb + (size_t)bh * DH * S_LEN;   // [d][s] row-major

  for (int kb = 0; kb <= q0 + 15; kb += 32) {
    // scores: Q(16x64) x K^T -> two 16x16 C tiles (keys kb..kb+31)
    const bf16* kr0 = Kb + (size_t)(b * S_LEN + kb + laneLo) * E_DIM + h * DH;
    const bf16* kr1 = kr0 + (size_t)16 * E_DIM;
    v8f s0 = {}, s1 = {};
    s0 = wmma_bf16(aq0, cat8(ld8(kr0 + 16 * laneHi),      ld8(kr0 + 16 * laneHi + 8)),      s0);
    s0 = wmma_bf16(aq1, cat8(ld8(kr0 + 32 + 16 * laneHi), ld8(kr0 + 32 + 16 * laneHi + 8)), s0);
    s1 = wmma_bf16(aq0, cat8(ld8(kr1 + 16 * laneHi),      ld8(kr1 + 16 * laneHi + 8)),      s1);
    s1 = wmma_bf16(aq1, cat8(ld8(kr1 + 32 + 16 * laneHi), ld8(kr1 + 32 + 16 * laneHi + 8)), s1);

    // hoist PV B-operands: their load latency hides under the softmax VALU
    v16bf bv[4];
#pragma unroll
    for (int t = 0; t < 4; ++t) {
      const bf16* vr = vbase + (size_t)(t * 16 + laneLo) * S_LEN + kb + 8 * laneHi;
      bv[t] = cat8(ld8(vr), ld8(vr + 16));
    }

    // online softmax per row (rows live at vgpr i + 8*laneHi; N across lanes)
#pragma unroll
    for (int i = 0; i < 8; ++i) {
      const int q = q0 + i + 8 * laneHi;
      const int n = kb + laneLo;
      float v0 = s0[i] * 0.125f; if (n > q)      v0 = -3.0e38f;
      float v1 = s1[i] * 0.125f; if (n + 16 > q) v1 = -3.0e38f;
      const float mx   = half_reduce_max(fmaxf(v0, v1));
      const float mnew = fmaxf(mrow[i], mx);
      const float r    = __expf(mrow[i] - mnew);
      const float p0   = __expf(v0 - mnew);
      const float p1   = __expf(v1 - mnew);
      lrow[i] = lrow[i] * r + half_reduce_sum(p0 + p1);
      mrow[i] = mnew;
      o[0][i] *= r; o[1][i] *= r; o[2][i] *= r; o[3][i] *= r;
      const int row = i + 8 * laneHi;
      P[row * 32 + laneLo]      = (bf16)p0;    // C-layout -> LDS row-major
      P[row * 32 + 16 + laneLo] = (bf16)p1;
    }
    // wave-local LDS RAW fence (CDNA5 split counter); no block barrier:
    // waves in this block have divergent causal trip counts.
    asm volatile("s_wait_dscnt 0" ::: "memory");

    // re-read P in A-operand layout (row = laneLo, K interleave via laneHi)
    const v16bf ap = cat8(ld8(P + laneLo * 32 + 8 * laneHi),
                          ld8(P + laneLo * 32 + 16 + 8 * laneHi));
    // O += P(16x32) x K_block(32x64)
#pragma unroll
    for (int t = 0; t < 4; ++t)
      o[t] = wmma_bf16(ap, bv[t], o[t]);
    asm volatile("" ::: "memory");
  }

#pragma unroll
  for (int i = 0; i < 8; ++i) {
    const float inv = 1.0f / lrow[i];
    bf16* orow = Ob + (size_t)(b * S_LEN + q0 + i + 8 * laneHi) * E_DIM
                    + h * DH + laneLo;
    orow[0]  = (bf16)(o[0][i] * inv);
    orow[16] = (bf16)(o[1][i] * inv);
    orow[32] = (bf16)(o[2][i] * inv);
    orow[48] = (bf16)(o[3][i] * inv);
  }
}

// ------------------------------- launcher ----------------------------------
extern "C" void kernel_launch(void* const* d_in, const int* in_sizes, int n_in,
                              void* d_out, int out_size, void* d_ws,
                              size_t ws_size, hipStream_t stream) {
  (void)in_sizes; (void)n_in; (void)out_size; (void)ws_size;
  const float* x     = (const float*)d_in[0];
  const float* Wq    = (const float*)d_in[1];
  const float* Wk    = (const float*)d_in[2];
  const float* Wproj = (const float*)d_in[3];
  const float* bproj = (const float*)d_in[4];
  const float* ln1g  = (const float*)d_in[5];
  const float* ln1b  = (const float*)d_in[6];
  const float* ln2g  = (const float*)d_in[7];
  const float* ln2b  = (const float*)d_in[8];
  const float* W1    = (const float*)d_in[9];
  const float* b1    = (const float*)d_in[10];
  const float* W2    = (const float*)d_in[11];
  const float* b2    = (const float*)d_in[12];
  float* out = (float*)d_out;

  char* ws = (char*)d_ws;
  size_t off = 0;
  auto alloc = [&](size_t bytes) {
    char* p = ws + off;
    off += (bytes + 255) & ~(size_t)255;
    return p;
  };
  bf16* hbf    = (bf16*)alloc((size_t)M_ROWS * E_DIM  * 2);  //  8 MB
  bf16* WqT    = (bf16*)alloc((size_t)E_DIM  * E_DIM  * 2);  //  2 MB
  bf16* WkT    = (bf16*)alloc((size_t)E_DIM  * E_DIM  * 2);  //  2 MB
  bf16* WprojT = (bf16*)alloc((size_t)E_DIM  * E_DIM  * 2);  //  2 MB
  bf16* W1T    = (bf16*)alloc((size_t)E_DIM  * FF_DIM * 2);  //  8 MB
  bf16* W2T    = (bf16*)alloc((size_t)FF_DIM * E_DIM  * 2);  //  8 MB
  bf16* Qb     = (bf16*)alloc((size_t)M_ROWS * E_DIM  * 2);  //  8 MB
  bf16* Kb     = (bf16*)alloc((size_t)M_ROWS * E_DIM  * 2);  //  8 MB
  bf16* Ktb    = (bf16*)alloc((size_t)M_ROWS * E_DIM  * 2);  //  8 MB
  bf16* attnb  = (bf16*)alloc((size_t)M_ROWS * E_DIM  * 2);  //  8 MB
  float* x2    = (float*)alloc((size_t)M_ROWS * E_DIM * 4);  // 16 MB
  bf16* h2bf   = (bf16*)alloc((size_t)M_ROWS * E_DIM  * 2);  //  8 MB
  bf16* ffbf   = (bf16*)alloc((size_t)M_ROWS * FF_DIM * 2);  // 32 MB

  // weight conversion / transpose (bf16, BT row-major-by-output-column)
  cvt_qkw_kernel<<<4096, 256, 0, stream>>>(Wq, WqT);
  cvt_qkw_kernel<<<4096, 256, 0, stream>>>(Wk, WkT);
  cvt_T_kernel<<<4096,  256, 0, stream>>>(Wproj, WprojT, E_DIM, E_DIM);
  cvt_T_kernel<<<16384, 256, 0, stream>>>(W1, W1T, E_DIM, FF_DIM);
  cvt_T_kernel<<<16384, 256, 0, stream>>>(W2, W2T, FF_DIM, E_DIM);

  // LN1: x -> hbf (bf16)
  ln_kernel<<<M_ROWS, 256, 0, stream>>>(x, ln1g, ln1b, hbf);

  // Q = h @ Wq ; K = h @ Wk (K also stored transposed [b,h,d,s])
  // tiles = (M/32)*(N/64); 8 waves per block
  gemm_bf16_kernel<<<256, 256, 0, stream>>>(hbf, WqT, nullptr, nullptr,
      nullptr, Qb, nullptr, M_ROWS, E_DIM, E_DIM, 0);
  gemm_bf16_kernel<<<256, 256, 0, stream>>>(hbf, WkT, nullptr, nullptr,
      nullptr, Kb, Ktb, M_ROWS, E_DIM, E_DIM, 0);

  // causal attention (V == K), out -> attnb [b,s,h*Dh] bf16
  attn_kernel<<<1024, 128, 0, stream>>>(Qb, Kb, Ktb, attnb);

  // x2 = x + attn @ Wproj + bproj   (fp32)
  gemm_bf16_kernel<<<256, 256, 0, stream>>>(attnb, WprojT, bproj, x,
      x2, nullptr, nullptr, M_ROWS, E_DIM, E_DIM, 0);

  // LN2: x2 -> h2bf
  ln_kernel<<<M_ROWS, 256, 0, stream>>>(x2, ln2g, ln2b, h2bf);

  // ff = relu(h2 @ W1 + b1)  (bf16)
  gemm_bf16_kernel<<<1024, 256, 0, stream>>>(h2bf, W1T, b1, nullptr,
      nullptr, ffbf, nullptr, M_ROWS, FF_DIM, E_DIM, 1);

  // out = x2 + ff @ W2 + b2  (fp32)
  gemm_bf16_kernel<<<256, 256, 0, stream>>>(ffbf, W2T, b2, x2,
      out, nullptr, nullptr, M_ROWS, E_DIM, FF_DIM, 0);
}
